// CapsuleLayer_56633438765080
// MI455X (gfx1250) — compile-verified, matrix operand order checked by reading, not run on hardware
//
#include <hip/hip_runtime.h>

// ---- problem constants ----
#define Bsz 128
#define Isz 1152
#define Dsz 8
#define Csz 32
#define Esz 16
#define Ksz (Isz*Dsz)          // 9216 contraction length per capsule

typedef __attribute__((ext_vector_type(16))) _Float16 v16h;
typedef __attribute__((ext_vector_type(8)))  _Float16 v8h;
typedef __attribute__((ext_vector_type(8)))  float    v8f;

union Frag { v16h v; v8h h[2]; };

// ---------------------------------------------------------------------------
// Prep: WtT[c][e][i*8+d] = (f16) W[i][c][d][e]   (one-time per launch)
// ---------------------------------------------------------------------------
__global__ __launch_bounds__(256) void prep_wt(const float* __restrict__ W,
                                               _Float16* __restrict__ wtT) {
  int o  = blockIdx.x * 256 + threadIdx.x;   // 0 .. C*E*K-1
  int k  = o % Ksz;
  int ce = o / Ksz;
  int e  = ce & 15;
  int c  = ce >> 4;
  int i  = k >> 3, d = k & 7;
  wtT[o] = (_Float16)W[((i * Csz + c) * Dsz + d) * Esz + e];
}

// ---------------------------------------------------------------------------
// Prep: xh = (f16) x   (one-time per launch)
// ---------------------------------------------------------------------------
__global__ __launch_bounds__(256) void prep_xh(const float* __restrict__ x,
                                               _Float16* __restrict__ xh) {
  int o = blockIdx.x * 256 + threadIdx.x;    // 0 .. B*I*D-1
  xh[o] = (_Float16)x[o];
}

// ---------------------------------------------------------------------------
// Softmax over C per (b,i) row:  cco = softmax(blog, axis=C)  (f16 out)
// one wave per row, lane == capsule index
// ---------------------------------------------------------------------------
__global__ __launch_bounds__(256) void softmax_k(const float* __restrict__ blog,
                                                 _Float16* __restrict__ cco) {
  int wave = threadIdx.x >> 5, lane = threadIdx.x & 31;
  int row  = blockIdx.x * 8 + wave;          // 0 .. B*I-1
  float v  = blog[row * Csz + lane];
  float mx = v;
  for (int m = 16; m >= 1; m >>= 1) mx = fmaxf(mx, __shfl_xor(mx, m, 32));
  float ex = __expf(v - mx);
  float sm = ex;
  for (int m = 16; m >= 1; m >>= 1) sm += __shfl_xor(sm, m, 32);
  cco[row * Csz + lane] = (_Float16)(ex / sm);
}

// ---------------------------------------------------------------------------
// s-pass (WMMA): per block (c, b_tile), per wave a K-chunk of 1152.
//   A[b, i*8+d] = cco[b,i,c] * xh[b,i,d]  (f16 multiply, wave-private LDS)
//   B[k, e]     = WtT[c][e][k]            (global, L2-resident)
// Pipelining: B-fragment loads issued before the A-build; A-build gathers all
// inputs into register arrays first (clause-able loads, partial waits), then
// does pure v_pk_mul_f16 + ds_store_b128. No block barriers: slab is
// wave-private and per-wave LDS ops are in-order (DScnt).
// ---------------------------------------------------------------------------
__global__ __launch_bounds__(256) void s_pass(const _Float16* __restrict__ xh,
                                              const _Float16* __restrict__ cco,
                                              const _Float16* __restrict__ wtT,
                                              float*          __restrict__ ppart) {
  __shared__ _Float16 aslab[8][16 * 128];    // per-wave A slab: 16 b x 128 k (4 KB) => 32 KB
  const int tid  = threadIdx.x;
  const int wave = tid >> 5, lane = tid & 31;
  const int c    = blockIdx.x;               // capsule
  const int bt   = blockIdx.y;               // batch tile (16 batches)
  const int half = lane >> 4, l = lane & 15;

  const int kbase_wave = wave * (Ksz / 8);   // 1152 k per wave, 9 slabs of 128
  v8f acc = {};

  // B row for this lane: e == l (lanes 16..31 use same e, different k-halves)
  const _Float16* wrow = wtT + (size_t)(c * Esz + l) * Ksz;
  _Float16* myslab = &aslab[wave][0];

  for (int slab = 0; slab < 9; ++slab) {
    const int kslab = kbase_wave + slab * 128;   // global k at slab start
    const int ibase = kslab >> 3;                // 16 i's per slab

    // ---- issue B-fragment loads for all 4 k-steps (long-latency, overlap) ----
    Frag bf[4];
#pragma unroll
    for (int s = 0; s < 4; ++s) {
      int k0 = s * 32 + half * 8;
      bf[s].h[0] = *(const v8h*)(wrow + (kslab + k0));
      bf[s].h[1] = *(const v8h*)(wrow + (kslab + k0 + 16));
    }

    // ---- A-build: gather all inputs first (batched loads) ----
    _Float16 cf[8];
    v8h      xv[8];
#pragma unroll
    for (int p = 0; p < 8; ++p) {
      int pr = lane + 32 * p;                    // 0..255, bijective
      int b  = pr & 15;
      int il = pr >> 4;
      int gb = bt * 16 + b;
      int gi = ibase + il;
      cf[p] = cco[(gb * Isz + gi) * Csz + c];
      xv[p] = *(const v8h*)(xh + ((size_t)gb * Isz + gi) * Dsz);
    }
    // ---- pure f16 multiply + store ----
#pragma unroll
    for (int p = 0; p < 8; ++p) {
      int pr = lane + 32 * p;
      int b  = pr & 15;
      int il = pr >> 4;
      v8h hv = xv[p] * cf[p];                    // splat scalar, v_pk_mul_f16
      *(v8h*)&myslab[b * 128 + il * 8] = hv;
    }

    // ---- load all 4 A fragments (in-order with the stores above) ----
    Frag af[4];
#pragma unroll
    for (int s = 0; s < 4; ++s) {
      int k0 = s * 32 + half * 8;
      af[s].h[0] = *(const v8h*)&myslab[l * 128 + k0];
      af[s].h[1] = *(const v8h*)&myslab[l * 128 + k0 + 16];
    }

    // ---- WMMA chain ----
#pragma unroll
    for (int s = 0; s < 4; ++s) {
      acc = __builtin_amdgcn_wmma_f32_16x16x32_f16(
          /*neg_a=*/false, af[s].v, /*neg_b=*/false, bf[s].v,
          /*c_mod=*/(short)0, acc, /*reuse_a=*/false, /*reuse_b=*/false);
    }
  }

  // write 16x16 wave partial: layout [bt][c][wave][m][n]
  float* pp = ppart + (size_t)(((bt * Csz + c) * 8 + wave) * 256);
#pragma unroll
  for (int r = 0; r < 8; ++r) pp[(r + half * 8) * 16 + l] = acc[r];
}

// ---------------------------------------------------------------------------
// Reduce wave partials, squash; last iteration also writes d_out.
// thread t -> (b, c, e); 16 consecutive lanes share a (b,c) row.
// ---------------------------------------------------------------------------
__global__ __launch_bounds__(256) void squash_k(const float* __restrict__ ppart,
                                                float* __restrict__ sout,
                                                float* __restrict__ dout,
                                                int last) {
  int t   = blockIdx.x * 256 + threadIdx.x;   // 0 .. B*C*E-1
  int e   = t & 15;
  int row = t >> 4;                           // b*C + c
  int c   = row & 31, b = row >> 5;
  int bt  = b >> 4,  m = b & 15;
  const float* base = ppart + (size_t)((bt * Csz + c) * 8) * 256 + m * 16 + e;
  float s = 0.f;
#pragma unroll
  for (int w = 0; w < 8; ++w) s += base[w * 256];
  float s2 = s * s;
  for (int msk = 8; msk >= 1; msk >>= 1) s2 += __shfl_xor(s2, msk, 16);
  float scale = s2 / (1.0f + s2) / sqrtf(s2);    // reference: no epsilon
  float o = scale * s;
  sout[t] = o;
  if (last) dout[t] = o;
}

// ---------------------------------------------------------------------------
// Logit update: blog[b,i,c] += sum_{d,e} x[b,i,d] * W[i,c,d,e] * out[b,c,e]
// block = (c, b_tile, i_tile of 32). Exclusive ownership -> plain RMW.
// ---------------------------------------------------------------------------
__global__ __launch_bounds__(256) void bupdate_k(const float*    __restrict__ x,
                                                 const _Float16* __restrict__ wtT,
                                                 const float*    __restrict__ sout,
                                                 float*          __restrict__ blog) {
  __shared__ float    xs[16 * 256];    // [b][i_local*8+d]  16 KB
  __shared__ _Float16 wt[16 * 256];    // [e][i_local*8+d]   8 KB
  __shared__ float    ot[256];         // [b][e]             1 KB
  const int tid = threadIdx.x;
  const int c = blockIdx.x, bt = blockIdx.y, it = blockIdx.z;
  const int i_base = it * 32;

#pragma unroll 4
  for (int j = 0; j < 16; ++j) {       // x tile: coalesced (256 floats / b row)
    int idx = tid + 256 * j;
    int b = idx >> 8, kl = idx & 255;
    xs[idx] = x[((size_t)(bt * 16 + b) * Isz + i_base) * Dsz + kl];
  }
#pragma unroll 4
  for (int j = 0; j < 16; ++j) {       // W tile: coalesced f16
    int idx = tid + 256 * j;
    int e = idx >> 8, kl = idx & 255;
    wt[idx] = wtT[(size_t)(c * Esz + e) * Ksz + i_base * Dsz + kl];
  }
  { int b = tid >> 4, e = tid & 15;
    ot[tid] = sout[((bt * 16 + b) * Csz + c) * Esz + e]; }
  __syncthreads();

  for (int pp = 0; pp < 2; ++pp) {
    int p  = tid + 256 * pp;           // 0..511 -> (b, i_local)
    int b  = p >> 5;
    int il = p & 31;
    float4 xa = *(const float4*)&xs[b * 256 + il * 8];
    float4 xb = *(const float4*)&xs[b * 256 + il * 8 + 4];
    float gd[8];
#pragma unroll
    for (int d = 0; d < 8; ++d) gd[d] = 0.f;
#pragma unroll
    for (int e = 0; e < 16; ++e) {
      v8h wv = *(const v8h*)&wt[e * 256 + il * 8];
      float o = ot[b * 16 + e];
#pragma unroll
      for (int d = 0; d < 8; ++d) gd[d] += (float)wv[d] * o;
    }
    float acc = xa.x * gd[0] + xa.y * gd[1] + xa.z * gd[2] + xa.w * gd[3]
              + xb.x * gd[4] + xb.y * gd[5] + xb.z * gd[6] + xb.w * gd[7];
    size_t bi = ((size_t)(bt * 16 + b) * Isz + i_base + il) * Csz + c;
    blog[bi] += acc;
  }
}

// ---------------------------------------------------------------------------
// workspace layout (bytes)
// ---------------------------------------------------------------------------
#define WT_OFF   0u                    // f16  C*E*K      =  9,437,184 B
#define XH_OFF   9437184u              // f16  B*I*D      =  2,359,296 B
#define CC_OFF   11796480u             // f16  B*I*C      =  9,437,184 B
#define BL_OFF   21233664u             // f32  B*I*C      = 18,874,368 B
#define PP_OFF   40108032u             // f32  8*32*8*256 =  2,097,152 B
#define SO_OFF   42205184u             // f32  B*C*E      =    262,144 B

extern "C" void kernel_launch(void* const* d_in, const int* in_sizes, int n_in,
                              void* d_out, int out_size, void* d_ws, size_t ws_size,
                              hipStream_t stream) {
  const float* x = (const float*)d_in[0];   // [B, I, D]
  const float* W = (const float*)d_in[1];   // [I, C, D, E]
  float* out = (float*)d_out;               // [B, C, E]
  char* ws = (char*)d_ws;

  _Float16* wtT  = (_Float16*)(ws + WT_OFF);
  _Float16* xh   = (_Float16*)(ws + XH_OFF);
  _Float16* cco  = (_Float16*)(ws + CC_OFF);
  float*    blog = (float*)   (ws + BL_OFF);
  float*    ppart= (float*)   (ws + PP_OFF);
  float*    sout = (float*)   (ws + SO_OFF);

  (void)in_sizes; (void)n_in; (void)out_size; (void)ws_size;

  hipMemsetAsync(blog, 0, (size_t)Bsz * Isz * Csz * sizeof(float), stream);
  prep_wt<<<(Csz * Esz * Ksz) / 256, 256, 0, stream>>>(W, wtT);
  prep_xh<<<(Bsz * Isz * Dsz) / 256, 256, 0, stream>>>(x, xh);

  for (int r = 0; r < 3; ++r) {
    softmax_k<<<(Bsz * Isz) / 8, 256, 0, stream>>>(blog, cco);
    s_pass<<<dim3(Csz, Bsz / 16), 256, 0, stream>>>(xh, cco, wtT, ppart);
    squash_k<<<(Bsz * Csz * Esz) / 256, 256, 0, stream>>>(ppart, sout, out, r == 2);
    if (r < 2)
      bupdate_k<<<dim3(Csz, Bsz / 16, Isz / 32), 256, 0, stream>>>(x, wtT, sout, blog);
  }
}
